// PNANet_65094524338706
// MI455X (gfx1250) — matrix operand order repaired and checked by reading, not compiled, not run on hardware
//
#include <hip/hip_runtime.h>
#include <hip/hip_bf16.h>
#include <stdint.h>

// ---------------- problem constants ----------------
#define NN 50000
#define EE 400000
#define GG 64
#define DD 100
#define LL 2
#define PD 128            // padded f16 feature stride
#define NT_N 3125         // NN/16 node tiles
#define NT_E 25000        // EE/16 edge tiles
#define KP_E 128          // edge-enc GEMM K pad
#define KP_P 384          // pre-NN GEMM K pad (3 segs * 128)
#define KP_Q 1664         // post-NN GEMM K pad (13 segs * 128)
#define KP_L 128          // lin GEMM K pad
#define AVGLOG 2.19722457733621938f
#define NEG_SLOPE 0.01f

typedef __attribute__((ext_vector_type(16))) _Float16 v16h;
typedef __attribute__((ext_vector_type(8)))  float    v8f;
typedef __attribute__((ext_vector_type(2)))  _Float16 h2;

__device__ inline v8f zero8() {
  v8f z = {0.f,0.f,0.f,0.f,0.f,0.f,0.f,0.f};
  return z;
}

__device__ inline v8f wmma16(v16h a, v16h b, v8f c) {
  // D = A(16x32 f16) * B(32x16 f16) + C(16x16 f32)
  return __builtin_amdgcn_wmma_f32_16x16x32_f16(false, a, false, b, (short)0, c,
                                                false, false);
}

// A fragment (16x32 f16). rowbase points at this lane's row (row = lane&15).
// ISA layout: pair p -> K = (p/4)*16 + half*8 + (p%4)*2, half = lane>>4.
__device__ inline v16h load_a(const _Float16* rowbase, int k0) {
  const int half = (threadIdx.x >> 4) & 1;
  const uint32_t* p = (const uint32_t*)rowbase;
  v16h a;
#pragma unroll
  for (int q = 0; q < 8; ++q) {
    int k = k0 + ((q >> 2) << 4) + (half << 3) + ((q & 3) << 1);
    h2 t = __builtin_bit_cast(h2, p[k >> 1]);
    a[2*q]   = t.x;
    a[2*q+1] = t.y;
  }
  return a;
}

__device__ inline v16h load_a_scaled(const _Float16* rowbase, int k0, float sc) {
  const int half = (threadIdx.x >> 4) & 1;
  const uint32_t* p = (const uint32_t*)rowbase;
  const _Float16 hs = (_Float16)sc;
  v16h a;
#pragma unroll
  for (int q = 0; q < 8; ++q) {
    int k = k0 + ((q >> 2) << 4) + (half << 3) + ((q & 3) << 1);
    h2 t = __builtin_bit_cast(h2, p[k >> 1]);
    a[2*q]   = t.x * hs;
    a[2*q+1] = t.y * hs;
  }
  return a;
}

// B fragment (32x16 f16) from transposed weights Wt[n][Kpad]:
// lane covers column n = n0 + (lane&15); pair p -> K = (p/4)*16 + half*8 + (p%4)*2,
// half = lane>>4 (same order as load_a); 8 dword loads per lane.
__device__ inline v16h load_b(const _Float16* __restrict__ Wt, int Kpad,
                              int n0, int k0) {
  const int lane = threadIdx.x & 31;
  const int n  = n0 + (lane & 15);
  const int kb = k0 + ((lane >> 4) << 3);
  const uint32_t* p = (const uint32_t*)(Wt + (size_t)n * Kpad + kb);
  v16h b;
#pragma unroll
  for (int r = 0; r < 8; ++r) {
    h2 t = __builtin_bit_cast(h2, p[((r >> 2) << 3) + (r & 3)]);
    b[2*r]   = t.x;
    b[2*r+1] = t.y;
  }
  return b;
}

__device__ inline float lrelu(float v) { return v >= 0.f ? v : NEG_SLOPE * v; }

// float atomic max/min via sign-aware integer atomics
__device__ inline void atomicMaxF(float* a, float v) {
  if (v >= 0.f) atomicMax((int*)a, __float_as_int(v));
  else          atomicMin((unsigned int*)a, __float_as_uint(v));
}
__device__ inline void atomicMinF(float* a, float v) {
  if (v >= 0.f) atomicMin((int*)a, __float_as_int(v));
  else          atomicMax((unsigned int*)a, __float_as_uint(v));
}

// ---------------- encoders ----------------
__global__ void k_atom(const int* __restrict__ x, const float* __restrict__ tab,
                       float* __restrict__ h32, _Float16* __restrict__ h16) {
  const int n = blockIdx.x;
  const int t = threadIdx.x;
  if (t >= PD) return;
  float acc = 0.f;
  if (t < DD) {
    const int AOFF[9] = {0,119,124,136,148,158,164,170,172};
#pragma unroll
    for (int k = 0; k < 9; ++k)
      acc += tab[(size_t)(x[n*9+k] + AOFF[k]) * DD + t];
    h32[(size_t)n*DD + t] = acc;
  }
  h16[(size_t)n*PD + t] = (_Float16)(t < DD ? acc : 0.f);
}

__global__ void k_bond(const int* __restrict__ ea, const float* __restrict__ tab,
                       _Float16* __restrict__ e16) {
  const int e = blockIdx.x;
  const int t = threadIdx.x;
  if (t >= PD) return;
  float acc = 0.f;
  if (t < DD) {
    const int BOFF[3] = {0, 5, 11};
#pragma unroll
    for (int k = 0; k < 3; ++k)
      acc += tab[(size_t)(ea[e*3+k] + BOFF[k]) * DD + t];
  }
  e16[(size_t)e*PD + t] = (_Float16)(t < DD ? acc : 0.f);
}

// ---------------- weight repack (transpose + pad, fp32->fp16) ----------------
// Wt[n][Kpad] from W[Ko][DD] (simple: single K segment)
__global__ void k_cvt_simple(const float* __restrict__ W, int Ko,
                             _Float16* __restrict__ Wt, int Kpad) {
  const size_t total = (size_t)128 * Kpad;
  const size_t stride = (size_t)gridDim.x * blockDim.x;
  for (size_t i = (size_t)blockIdx.x*blockDim.x + threadIdx.x; i < total; i += stride) {
    int n = (int)(i / Kpad), k = (int)(i % Kpad);
    float v = (n < DD && k < Ko) ? W[(size_t)k*DD + n] : 0.f;
    Wt[i] = (_Float16)v;
  }
}
// Wt[n][S*128] from W[S*100][DD]; each 100-row segment padded to 128
__global__ void k_cvt_seg(const float* __restrict__ W, int S,
                          _Float16* __restrict__ Wt) {
  const int Kpad = S * 128;
  const size_t total = (size_t)128 * Kpad;
  const size_t stride = (size_t)gridDim.x * blockDim.x;
  for (size_t i = (size_t)blockIdx.x*blockDim.x + threadIdx.x; i < total; i += stride) {
    int n = (int)(i / Kpad), k = (int)(i % Kpad);
    int s = k >> 7, off = k & 127;
    float v = (n < DD && off < DD) ? W[(size_t)(s*DD + off)*DD + n] : 0.f;
    Wt[i] = (_Float16)v;
  }
}

// ---------------- per-layer init ----------------
__global__ void k_init_agg(float* asum, float* asum2, float* amn, float* amx,
                           float* cnt, float* bnsum, float* bnsum2) {
  const size_t total = (size_t)NN * DD;
  const size_t stride = (size_t)gridDim.x * blockDim.x;
  for (size_t i = (size_t)blockIdx.x*blockDim.x + threadIdx.x; i < total; i += stride) {
    asum[i]  = 0.f;
    asum2[i] = 0.f;
    amn[i]   =  __builtin_inff();
    amx[i]   = -__builtin_inff();
    if (i < NN)  cnt[i] = 0.f;
    if (i < 128) { bnsum[i] = 0.f; bnsum2[i] = 0.f; }
  }
}

// ---------------- edge-encoder GEMM: eenc = e @ eW + eb ----------------
__global__ void __launch_bounds__(64)
k_edgeenc(const _Float16* __restrict__ e16, const _Float16* __restrict__ Wt,
          const float* __restrict__ eb, _Float16* __restrict__ eenc) {
  const int wave = threadIdx.x >> 5, lane = threadIdx.x & 31;
  const int tile = blockIdx.x * 2 + wave;
  if (tile >= NT_E) return;
  const int e0 = tile << 4;
  const int row = lane & 15, half = lane >> 4, coln = lane & 15;
  const _Float16* rb = e16 + (size_t)(e0 + row) * PD;

  v8f c[7];
#pragma unroll
  for (int i = 0; i < 7; ++i) c[i] = zero8();

#pragma unroll
  for (int kt = 0; kt < 4; ++kt) {
    v16h a = load_a(rb, kt * 32);
#pragma unroll
    for (int nt = 0; nt < 7; ++nt)
      c[nt] = wmma16(a, load_b(Wt, KP_E, nt*16, kt*32), c[nt]);
  }
#pragma unroll
  for (int v = 0; v < 8; ++v) {
    const int r = v + half*8;
    _Float16* orow = eenc + (size_t)(e0 + r) * PD;
#pragma unroll
    for (int nt = 0; nt < 7; ++nt) {
      const int n = nt*16 + coln;
      float val = (n < DD) ? (c[nt][v] + eb[n]) : 0.f;
      orow[n] = (_Float16)val;
    }
  }
}

// ---------------- message GEMM + fused scatter aggregation ----------------
// m = [h[dst] | h[src] | eenc] @ pW + pb, then atomics into sum/sum2/min/max/cnt
__global__ void __launch_bounds__(64)
k_msg(const _Float16* __restrict__ h16, const _Float16* __restrict__ eenc,
      const int* __restrict__ src, const int* __restrict__ dst,
      const _Float16* __restrict__ Wt, const float* __restrict__ pb,
      float* __restrict__ asum, float* __restrict__ asum2,
      float* __restrict__ amn, float* __restrict__ amx, float* __restrict__ cnt) {
  const int wave = threadIdx.x >> 5, lane = threadIdx.x & 31;
  const int tile = blockIdx.x * 2 + wave;
  if (tile >= NT_E) return;
  const int e0 = tile << 4;
  const int row = lane & 15, half = lane >> 4, coln = lane & 15;

  const int di = dst[e0 + row];
  const int si = src[e0 + row];
  const _Float16* base0 = h16  + (size_t)di * PD;
  const _Float16* base1 = h16  + (size_t)si * PD;
  const _Float16* base2 = eenc + (size_t)(e0 + row) * PD;

  v8f c[7];
#pragma unroll
  for (int i = 0; i < 7; ++i) c[i] = zero8();

#pragma unroll
  for (int kt = 0; kt < 12; ++kt) {
    const int s = kt >> 2;
    const _Float16* rb = (s == 0) ? base0 : (s == 1 ? base1 : base2);
    v16h a = load_a(rb, (kt & 3) * 32);
#pragma unroll
    for (int nt = 0; nt < 7; ++nt)
      c[nt] = wmma16(a, load_b(Wt, KP_P, nt*16, kt*32), c[nt]);
  }

#pragma unroll
  for (int v = 0; v < 8; ++v) {
    const int r = v + half*8;
    const int d2 = dst[e0 + r];
    const size_t rb2 = (size_t)d2 * DD;
#pragma unroll
    for (int nt = 0; nt < 7; ++nt) {
      const int n = nt*16 + coln;
      if (n < DD) {
        float m = c[nt][v] + pb[n];
        size_t o = rb2 + n;
        atomicAdd(asum  + o, m);
        atomicAdd(asum2 + o, m * m);
        atomicMaxF(amx + o, m);
        atomicMinF(amn + o, m);
      }
    }
  }
  if (lane < 16) atomicAdd(cnt + di, 1.0f);
}

// ---------------- fused post-NN + lin GEMM + BN-stat epilogue ----------------
// A = [h | mean mn mx std | *amp | *att]  (13 segs x 128) built on the fly
__global__ void __launch_bounds__(64)
k_post(const _Float16* __restrict__ h16,
       const float* __restrict__ asum, const float* __restrict__ asum2,
       const float* __restrict__ amn, const float* __restrict__ amx,
       const float* __restrict__ cnt,
       const _Float16* __restrict__ qWt, const float* __restrict__ qb,
       const _Float16* __restrict__ lWt, const float* __restrict__ lb,
       float* __restrict__ out32, float* __restrict__ bnsum,
       float* __restrict__ bnsum2) {
  __shared__ _Float16 s_agg[2][4][16*PD];   // mean, mn, mx, std tiles (per wave)
  __shared__ _Float16 s_t[2][16*PD];        // post-NN output tile (per wave)
  __shared__ float s_amp[2][16], s_att[2][16];

  const int wave = threadIdx.x >> 5, lane = threadIdx.x & 31;
  const int tile = blockIdx.x * 2 + wave;
  if (tile >= NT_N) return;                  // wave-private LDS: no barrier needed
  const int n0 = tile << 4;
  const int row = lane & 15, half = lane >> 4, coln = lane & 15;

  // ---- stage aggregator stats into LDS (f16, zero-padded) ----
  for (int idx = lane; idx < 16*PD; idx += 32) {
    const int r = idx >> 7, cdx = idx & 127;
    float mean = 0.f, mn = 0.f, mx = 0.f, sd = 0.f;
    if (cdx < DD) {
      const int node = n0 + r;
      const float cv = cnt[node];
      const float c1 = fmaxf(cv, 1.f);
      const size_t o = (size_t)node * DD + cdx;
      const float s1 = asum[o], s2 = asum2[o];
      mean = s1 / c1;
      float var = s2 / c1 - mean * mean;
      var = var > 0.f ? var : 0.f;
      sd = sqrtf(var + 1e-5f);
      if (cv > 0.f) { mn = amn[o]; mx = amx[o]; }
    }
    s_agg[wave][0][idx] = (_Float16)mean;
    s_agg[wave][1][idx] = (_Float16)mn;
    s_agg[wave][2][idx] = (_Float16)mx;
    s_agg[wave][3][idx] = (_Float16)sd;
    s_t[wave][idx] = (_Float16)0.f;
  }
  if (lane < 16) {
    const float dlog = log1pf(fmaxf(cnt[n0 + lane], 1.f));
    s_amp[wave][lane] = dlog / AVGLOG;
    s_att[wave][lane] = AVGLOG / dlog;
  }

  // ---- post-NN GEMM over 52 K-tiles (13 segments x 4 tiles) ----
  v8f c[7];
#pragma unroll
  for (int i = 0; i < 7; ++i) c[i] = zero8();

  for (int kt = 0; kt < 52; ++kt) {
    const int s = kt >> 2, k0 = (kt & 3) * 32;
    v16h a;
    if (s == 0) {
      a = load_a(h16 + (size_t)(n0 + row) * PD, k0);
    } else {
      const int g = (s - 1) & 3, grp = (s - 1) >> 2;
      const _Float16* rb = &s_agg[wave][g][row * PD];
      if (grp == 0) a = load_a(rb, k0);
      else a = load_a_scaled(rb, k0,
                             grp == 1 ? s_amp[wave][row] : s_att[wave][row]);
    }
#pragma unroll
    for (int nt = 0; nt < 7; ++nt)
      c[nt] = wmma16(a, load_b(qWt, KP_Q, nt*16, kt*32), c[nt]);
  }

  // ---- write post-NN tile (+qb) to LDS as f16 for the lin GEMM ----
#pragma unroll
  for (int v = 0; v < 8; ++v) {
    const int r = v + half*8;
#pragma unroll
    for (int nt = 0; nt < 7; ++nt) {
      const int n = nt*16 + coln;
      float val = (n < DD) ? (c[nt][v] + qb[n]) : 0.f;
      s_t[wave][r*PD + n] = (_Float16)val;
    }
  }

  // ---- lin GEMM: out = t @ lW + lb ----
  v8f c2[7];
#pragma unroll
  for (int i = 0; i < 7; ++i) c2[i] = zero8();
#pragma unroll
  for (int kt = 0; kt < 4; ++kt) {
    v16h a = load_a(&s_t[wave][row * PD], kt * 32);
#pragma unroll
    for (int nt = 0; nt < 7; ++nt)
      c2[nt] = wmma16(a, load_b(lWt, KP_L, nt*16, kt*32), c2[nt]);
  }

  // ---- epilogue: store pre-BN output, wave-reduced BN-stat atomics ----
#pragma unroll
  for (int nt = 0; nt < 7; ++nt) {
    const int n = nt*16 + coln;
    float p = 0.f, p2 = 0.f;
#pragma unroll
    for (int v = 0; v < 8; ++v) {
      const int r = v + half*8;
      if (n < DD) {
        float o = c2[nt][v] + lb[n];
        out32[(size_t)(n0 + r) * DD + n] = o;
        p  += o;
        p2 += o * o;
      }
    }
    p  += __shfl_xor(p, 16, 32);
    p2 += __shfl_xor(p2, 16, 32);
    if (half == 0 && n < DD) {
      atomicAdd(bnsum  + n, p);
      atomicAdd(bnsum2 + n, p2);
    }
  }
}

// ---------------- BN finalize + apply ----------------
__global__ void k_bnfin(const float* __restrict__ bnsum,
                        const float* __restrict__ bnsum2,
                        const float* __restrict__ gamma,
                        const float* __restrict__ beta,
                        float* __restrict__ scale, float* __restrict__ shift) {
  const int t = threadIdx.x;
  if (t < DD) {
    const float mu  = bnsum[t]  / (float)NN;
    const float var = bnsum2[t] / (float)NN - mu * mu;
    const float inv = rsqrtf(var + 1e-5f);
    const float sc  = gamma[t] * inv;
    scale[t] = sc;
    shift[t] = beta[t] - mu * sc;
  }
}

__global__ void k_apply(const float* __restrict__ out32,
                        const float* __restrict__ scale,
                        const float* __restrict__ shift,
                        float* __restrict__ h32, _Float16* __restrict__ h16) {
  const size_t total = (size_t)NN * DD;
  const size_t stride = (size_t)gridDim.x * blockDim.x;
  for (size_t i = (size_t)blockIdx.x*blockDim.x + threadIdx.x; i < total; i += stride) {
    const int n = (int)(i % DD);
    const size_t node = i / DD;
    float v = lrelu(out32[i] * scale[n] + shift[n]);
    h32[i] = v;
    h16[node * PD + n] = (_Float16)v;
  }
}

// ---------------- pooling + head MLP ----------------
__global__ void k_pool(const float* __restrict__ h32, const int* __restrict__ batch,
                       float* __restrict__ gsum, float* __restrict__ gcnt) {
  const size_t total = (size_t)NN * DD;
  const size_t stride = (size_t)gridDim.x * blockDim.x;
  for (size_t i = (size_t)blockIdx.x*blockDim.x + threadIdx.x; i < total; i += stride) {
    const int n = (int)(i % DD);
    const size_t node = i / DD;
    const int b = batch[node];
    atomicAdd(gsum + (size_t)b * DD + n, h32[i]);
    if (n == 0) atomicAdd(gcnt + b, 1.0f);
  }
}

__global__ void k_mlp(const float* __restrict__ gsum, const float* __restrict__ gcnt,
                      const float* __restrict__ W1, const float* __restrict__ b1,
                      const float* __restrict__ W2, const float* __restrict__ b2,
                      const float* __restrict__ W3, const float* __restrict__ b3,
                      float* __restrict__ out) {
  const int g = threadIdx.x;
  if (g >= GG) return;
  const float c1 = fmaxf(gcnt[g], 1.f);
  float gv[DD];
  for (int d = 0; d < DD; ++d) gv[d] = gsum[(size_t)g*DD + d] / c1;
  float a1[40];
  for (int j = 0; j < 40; ++j) {
    float s = b1[j];
    for (int d = 0; d < DD; ++d) s += gv[d] * W1[d*40 + j];
    a1[j] = lrelu(s);
  }
  float a2[20];
  for (int j = 0; j < 20; ++j) {
    float s = b2[j];
    for (int d = 0; d < 40; ++d) s += a1[d] * W2[d*20 + j];
    a2[j] = lrelu(s);
  }
  float s = b3[0];
  for (int d = 0; d < 20; ++d) s += a2[d] * W3[d];
  out[g] = s;
}

// ---------------- host orchestration ----------------
extern "C" void kernel_launch(void* const* d_in, const int* in_sizes, int n_in,
                              void* d_out, int out_size, void* d_ws, size_t ws_size,
                              hipStream_t stream) {
  (void)in_sizes; (void)n_in; (void)out_size; (void)ws_size;
  const int*   x     = (const int*)d_in[0];
  const int*   ei    = (const int*)d_in[1];
  const int*   ea    = (const int*)d_in[2];
  const int*   batch = (const int*)d_in[3];
  const float* atab  = (const float*)d_in[4];
  const float* btab  = (const float*)d_in[5];
  const float* eW    = (const float*)d_in[6];
  const float* eb    = (const float*)d_in[7];
  const float* pW    = (const float*)d_in[8];
  const float* pb    = (const float*)d_in[9];
  const float* qW    = (const float*)d_in[10];
  const float* qb    = (const float*)d_in[11];
  const float* lW    = (const float*)d_in[12];
  const float* lb    = (const float*)d_in[13];
  const float* gam   = (const float*)d_in[14];
  const float* bet   = (const float*)d_in[15];
  const float* W1    = (const float*)d_in[16];
  const float* b1    = (const float*)d_in[17];
  const float* W2    = (const float*)d_in[18];
  const float* b2    = (const float*)d_in[19];
  const float* W3    = (const float*)d_in[20];
  const float* b3    = (const float*)d_in[21];
  float* out = (float*)d_out;

  char* w = (char*)d_ws;
  size_t off = 0;
  auto carve = [&](size_t bytes) -> char* {
    char* p = w + off;
    off += (bytes + 255) & ~(size_t)255;
    return p;
  };

  float*    h32   = (float*)   carve((size_t)NN*DD*4);
  float*    o32   = (float*)   carve((size_t)NN*DD*4);
  _Float16* h16   = (_Float16*)carve((size_t)NN*PD*2);
  _Float16* e16   = (_Float16*)carve((size_t)EE*PD*2);
  _Float16* ee16  = (_Float16*)carve((size_t)EE*PD*2);
  float*    asum  = (float*)   carve((size_t)NN*DD*4);
  float*    asum2 = (float*)   carve((size_t)NN*DD*4);
  float*    amn   = (float*)   carve((size_t)NN*DD*4);
  float*    amx   = (float*)   carve((size_t)NN*DD*4);
  float*    cntb  = (float*)   carve((size_t)NN*4);
  float*    bnsum = (float*)   carve(512);
  float*    bnsum2= (float*)   carve(512);
  float*    bnsc  = (float*)   carve(512);
  float*    bnsh  = (float*)   carve(512);
  float*    gsum  = (float*)   carve((size_t)GG*DD*4);
  float*    gcnt  = (float*)   carve((size_t)GG*4);
  _Float16* eWt   = (_Float16*)carve((size_t)LL*128*KP_E*2);
  _Float16* pWt   = (_Float16*)carve((size_t)LL*128*KP_P*2);
  _Float16* qWt   = (_Float16*)carve((size_t)LL*128*KP_Q*2);
  _Float16* lWt   = (_Float16*)carve((size_t)LL*128*KP_L*2);

  const int* srcI = ei;
  const int* dstI = ei + EE;

  // encoders + one-time pad zeroing for e_enc columns 112..127
  k_atom<<<NN, 128, 0, stream>>>(x, atab, h32, h16);
  k_bond<<<EE, 128, 0, stream>>>(ea, btab, e16);
  hipMemsetAsync(ee16, 0, (size_t)EE*PD*2, stream);

  // weight repack (transpose + pad to f16)
  for (int l = 0; l < LL; ++l) {
    k_cvt_simple<<<64, 256, 0, stream>>>(eW + (size_t)l*DD*DD, DD,
                                         eWt + (size_t)l*128*KP_E, KP_E);
    k_cvt_seg<<<192, 256, 0, stream>>>(pW + (size_t)l*3*DD*DD, 3,
                                       pWt + (size_t)l*128*KP_P);
    k_cvt_seg<<<832, 256, 0, stream>>>(qW + (size_t)l*13*DD*DD, 13,
                                       qWt + (size_t)l*128*KP_Q);
    k_cvt_simple<<<64, 256, 0, stream>>>(lW + (size_t)l*DD*DD, DD,
                                         lWt + (size_t)l*128*KP_L, KP_L);
  }

  for (int l = 0; l < LL; ++l) {
    k_init_agg<<<4096, 256, 0, stream>>>(asum, asum2, amn, amx, cntb, bnsum, bnsum2);
    k_edgeenc<<<NT_E/2, 64, 0, stream>>>(e16, eWt + (size_t)l*128*KP_E,
                                         eb + (size_t)l*DD, ee16);
    k_msg<<<NT_E/2, 64, 0, stream>>>(h16, ee16, srcI, dstI,
                                     pWt + (size_t)l*128*KP_P, pb + (size_t)l*DD,
                                     asum, asum2, amn, amx, cntb);
    k_post<<<(NT_N+1)/2, 64, 0, stream>>>(h16, asum, asum2, amn, amx, cntb,
                                          qWt + (size_t)l*128*KP_Q, qb + (size_t)l*DD,
                                          lWt + (size_t)l*128*KP_L, lb + (size_t)l*DD,
                                          o32, bnsum, bnsum2);
    k_bnfin<<<1, 128, 0, stream>>>(bnsum, bnsum2, gam + (size_t)l*DD,
                                   bet + (size_t)l*DD, bnsc, bnsh);
    k_apply<<<2048, 256, 0, stream>>>(o32, bnsc, bnsh, h32, h16);
  }

  hipMemsetAsync(gsum, 0, (size_t)GG*DD*4, stream);
  hipMemsetAsync(gcnt, 0, (size_t)GG*4, stream);
  k_pool<<<2048, 256, 0, stream>>>(h32, batch, gsum, gcnt);
  k_mlp<<<1, 64, 0, stream>>>(gsum, gcnt, W1, b1, W2, b2, W3, b3, out);
}
